// VQVAE_36618891166244
// MI455X (gfx1250) — compile-verified, matrix operand order checked by reading, not run on hardware
//
#include <hip/hip_runtime.h>
#include <hip/hip_bf16.h>
#include <stddef.h>

// ---------------- problem constants ----------------
#define BB      8192
#define IN_DIM  1024
#define HID     2048
#define KCODES  512
#define DD      64
#define LL      32
#define RTOT    (BB * LL)          // 262144 latent rows

typedef unsigned short u16;
typedef __attribute__((ext_vector_type(16))) __bf16 v16bf;
typedef __attribute__((ext_vector_type(8)))  float  v8f;

union Frag {
    uint4 q[2];
    v16bf v;
};

__device__ __forceinline__ u16 f2bf(float f) {
    union { float f; unsigned u; } v; v.f = f;
    unsigned u = v.u;
    u += 0x7FFFu + ((u >> 16) & 1u);          // round-to-nearest-even
    return (u16)(u >> 16);
}

// ---- CDNA5 async global->LDS copy (ASYNCcnt-tracked, no VGPR staging) ----
__device__ __forceinline__ void async_copy_b128(unsigned lds_byte_off, const void* gptr) {
    asm volatile("global_load_async_to_lds_b128 %0, %1, off"
                 :: "v"(lds_byte_off), "v"(gptr) : "memory");
}
__device__ __forceinline__ void wait_async0() {
    asm volatile("s_wait_asynccnt 0x0" ::: "memory");
}

// ---------------- conversion kernels ----------------
__global__ void cvt_f32_bf16(const float* __restrict__ in, u16* __restrict__ out, int n4) {
    int i = blockIdx.x * blockDim.x + threadIdx.x;
    if (i < n4) {
        float4 v = ((const float4*)in)[i];
        uint2 p;
        p.x = (unsigned)f2bf(v.x) | ((unsigned)f2bf(v.y) << 16);
        p.y = (unsigned)f2bf(v.z) | ((unsigned)f2bf(v.w) << 16);
        ((uint2*)out)[i] = p;
    }
}

// W [Kd, Nd] f32 row-major  ->  Wt [Nd, Kd] bf16 row-major
__global__ void cvt_transpose_bf16(const float* __restrict__ W, u16* __restrict__ Wt,
                                   int Kd, int Nd) {
    __shared__ float t[32][33];
    int nb = blockIdx.x * 32, kb = blockIdx.y * 32;
    for (int i = threadIdx.y; i < 32; i += 8)
        t[i][threadIdx.x] = W[(size_t)(kb + i) * Nd + nb + threadIdx.x];
    __syncthreads();
    for (int i = threadIdx.y; i < 32; i += 8)
        Wt[(size_t)(nb + i) * Kd + kb + threadIdx.x] = f2bf(t[threadIdx.x][i]);
}

__global__ void emb_norm_kernel(const float* __restrict__ emb, float* __restrict__ nrm) {
    int k = blockIdx.x * blockDim.x + threadIdx.x;
    if (k < KCODES) {
        float s = 0.f;
        for (int d = 0; d < DD; ++d) { float v = emb[k * DD + d]; s = fmaf(v, v, s); }
        nrm[k] = s;
    }
}

// ---------------- WMMA GEMM with async double-buffered LDS pipeline ----------------
// C[M,N] = act( A[M,K](bf16) @ Bt[N,K]^T(bf16) + bias[N] )
// block tile 128x128, K-step 32, 256 threads (8 waves), wave w owns rows [w*16, w*16+16)
#define TM   128
#define TN   128
#define TK   32
#define LDSW 40   // LDS row stride in bf16 (32 data + 8 pad) -> conflict-free b128 lane loads

__global__ __launch_bounds__(256) void gemm_bf16(
        const u16* __restrict__ A, const u16* __restrict__ Bt,
        const float* __restrict__ bias,
        float* __restrict__ Cf, u16* __restrict__ Cb,
        int M, int N, int K, int relu)
{
    __shared__ __align__(16) u16 As[2][TM * LDSW];
    __shared__ __align__(16) u16 Bs[2][TN * LDSW];

    const int tid  = threadIdx.x;
    const int lane = tid & 31;
    const int wave = tid >> 5;
    const int hf   = lane >> 4;        // which 16-lane half of the wave
    const int l15  = lane & 15;
    const int m0   = blockIdx.y * TM;
    const int n0   = blockIdx.x * TN;

    v8f acc[8];
#pragma unroll
    for (int i = 0; i < 8; ++i) acc[i] = (v8f){0.f,0.f,0.f,0.f,0.f,0.f,0.f,0.f};

    const int row = tid >> 1;          // 0..127
    const int seg = (tid & 1) * 16;    // 0 or 16 (bf16 units)
    const u16* gA = A  + (size_t)(m0 + row) * K + seg;   // + k0 per tile
    const u16* gB = Bt + (size_t)(n0 + row) * K + seg;

    // issue async copies of K-slab starting at k0 into buffer `buf`
    auto issue = [&](int k0, int buf) {
        unsigned la = (unsigned)(size_t)(As[buf] + row * LDSW + seg);
        unsigned lb = (unsigned)(size_t)(Bs[buf] + row * LDSW + seg);
        async_copy_b128(la,      gA + k0);
        async_copy_b128(la + 16, gA + k0 + 8);
        async_copy_b128(lb,      gB + k0);
        async_copy_b128(lb + 16, gB + k0 + 8);
        if (k0 + TK < K) {             // pull the slab after next toward L2
            __builtin_prefetch(gA + k0 + TK, 0, 1);
            __builtin_prefetch(gB + k0 + TK, 0, 1);
        }
    };

    const int nk = K / TK;
    issue(0, 0);

    for (int ki = 0; ki < nk; ++ki) {
        const int buf = ki & 1;
        wait_async0();                 // my copies into `buf` landed
        __syncthreads();               // everyone's copies landed; prev reads done
        if (ki + 1 < nk) issue((ki + 1) * TK, buf ^ 1);   // overlap with compute

        // ---- A fragment: row (wave*16 + l15), k-pattern per 16-bit A layout ----
        const u16* arow = As[buf] + (wave * 16 + l15) * LDSW;
        Frag a;
        a.q[0] = *(const uint4*)(arow + 8 * hf);        // elems 0..7  : k = 8h..8h+7
        a.q[1] = *(const uint4*)(arow + 16 + 8 * hf);   // elems 8..15 : k = 16+8h..

#pragma unroll
        for (int nt = 0; nt < 8; ++nt) {
            const u16* brow = Bs[buf] + (nt * 16 + l15) * LDSW;  // Bt row-major == B^T
            Frag b;
            b.q[0] = *(const uint4*)(brow + 8 * hf);
            b.q[1] = *(const uint4*)(brow + 16 + 8 * hf);
            acc[nt] = __builtin_amdgcn_wmma_f32_16x16x32_bf16(
                          false, a.v, false, b.v, (short)0, acc[nt], false, false);
        }
    }

    // ---- epilogue: bias + optional ReLU, dual-precision store ----
#pragma unroll
    for (int nt = 0; nt < 8; ++nt) {
        int col = n0 + nt * 16 + l15;
        float bv = bias[col];
#pragma unroll
        for (int r = 0; r < 8; ++r) {
            int rr = m0 + wave * 16 + r + 8 * hf;
            float v = acc[nt][r] + bv;
            if (relu) v = fmaxf(v, 0.f);
            if (Cf) Cf[(size_t)rr * N + col] = v;
            if (Cb) Cb[(size_t)rr * N + col] = f2bf(v);
        }
    }
}

// ---------------- VQ: nearest codebook entry via WMMA scores + argmin ----------------
// scores tile = z(16x64) @ emb^T(64x16); argmin over ||e||^2 - 2*z.e (||z||^2 constant)
__global__ __launch_bounds__(256) void vq_kernel(
        const u16*  __restrict__ zb,     // [RTOT, 64] bf16
        const u16*  __restrict__ embb,   // [512, 64]  bf16
        const float* __restrict__ embf,  // [512, 64]  f32
        const float* __restrict__ enorm, // [512]
        int*   __restrict__ idx_out,     // [RTOT]
        float* __restrict__ zq_f,        // [RTOT, 64]
        u16*   __restrict__ zq_b)        // [RTOT, 64] bf16
{
    #define SEP 72                        // padded LDS row stride (conflict-free)
    __shared__ __align__(16) u16   se[KCODES * SEP];
    __shared__ float sn[KCODES];

    const int tid = threadIdx.x;
    // stage full bf16 codebook (64KB) into LDS via async copies; norms via VALU loads
    for (int i = tid; i < KCODES * (DD / 8); i += 256) {   // 4096 b128 packets
        int r = i >> 3, q = i & 7;
        unsigned dst = (unsigned)(size_t)(se + r * SEP) + q * 16;
        async_copy_b128(dst, ((const uint4*)embb) + i);
    }
    for (int i = tid; i < KCODES; i += 256) sn[i] = enorm[i];
    wait_async0();
    __syncthreads();

    const int lane = tid & 31, wave = tid >> 5;
    const int hf = lane >> 4, l15 = lane & 15;
    const size_t rb = (size_t)blockIdx.x * 128 + wave * 16;

    // A fragments for this wave's 16 rows, K = 0..31 and 32..63
    const u16* zr = zb + (rb + l15) * DD;
    Frag a0, a1;
    a0.q[0] = *(const uint4*)(zr + 8 * hf);
    a0.q[1] = *(const uint4*)(zr + 16 + 8 * hf);
    a1.q[0] = *(const uint4*)(zr + 32 + 8 * hf);
    a1.q[1] = *(const uint4*)(zr + 48 + 8 * hf);

    float best[8]; int bidx[8];
#pragma unroll
    for (int r = 0; r < 8; ++r) { best[r] = 3.4e38f; bidx[r] = 0; }

    for (int ct = 0; ct < KCODES / 16; ++ct) {
        const u16* er = se + (ct * 16 + l15) * SEP;   // emb row-major == B^T
        Frag b0, b1;
        b0.q[0] = *(const uint4*)(er + 8 * hf);
        b0.q[1] = *(const uint4*)(er + 16 + 8 * hf);
        b1.q[0] = *(const uint4*)(er + 32 + 8 * hf);
        b1.q[1] = *(const uint4*)(er + 48 + 8 * hf);
        v8f acc = (v8f){0.f,0.f,0.f,0.f,0.f,0.f,0.f,0.f};
        acc = __builtin_amdgcn_wmma_f32_16x16x32_bf16(false, a0.v, false, b0.v,
                                                      (short)0, acc, false, false);
        acc = __builtin_amdgcn_wmma_f32_16x16x32_bf16(false, a1.v, false, b1.v,
                                                      (short)0, acc, false, false);
        const int code = ct * 16 + l15;
        const float en = sn[code];
#pragma unroll
        for (int r = 0; r < 8; ++r) {
            float s = fmaf(-2.f, acc[r], en);
            if (s < best[r]) { best[r] = s; bidx[r] = code; }
        }
    }

    // argmin butterfly within each 16-lane half (wave32); first-index tiebreak
#pragma unroll
    for (int off = 8; off >= 1; off >>= 1) {
#pragma unroll
        for (int r = 0; r < 8; ++r) {
            float ob = __shfl_xor(best[r], off, 32);
            int   oi = __shfl_xor(bidx[r], off, 32);
            if (ob < best[r] || (ob == best[r] && oi < bidx[r])) { best[r] = ob; bidx[r] = oi; }
        }
    }

#pragma unroll
    for (int r = 0; r < 8; ++r) {
        size_t rr = rb + r + 8 * hf;
        int w = bidx[r];
        if (l15 == 0) idx_out[rr] = w;
        float4 ev = *(const float4*)(embf + (size_t)w * DD + l15 * 4);
        *(float4*)(zq_f + rr * DD + l15 * 4) = ev;
        uint2 p;
        p.x = (unsigned)f2bf(ev.x) | ((unsigned)f2bf(ev.y) << 16);
        p.y = (unsigned)f2bf(ev.z) | ((unsigned)f2bf(ev.w) << 16);
        *(uint2*)(zq_b + rr * DD + l15 * 4) = p;
    }
}

// ---------------- host-side orchestration ----------------
static inline size_t alignup(size_t x) { return (x + 255) & ~(size_t)255; }

extern "C" void kernel_launch(void* const* d_in, const int* in_sizes, int n_in,
                              void* d_out, int out_size, void* d_ws, size_t ws_size,
                              hipStream_t stream) {
    const float* x   = (const float*)d_in[0];
    const float* W1  = (const float*)d_in[1];
    const float* b1  = (const float*)d_in[2];
    const float* W2  = (const float*)d_in[3];
    const float* b2  = (const float*)d_in[4];
    const float* emb = (const float*)d_in[5];
    const float* W3  = (const float*)d_in[6];
    const float* b3  = (const float*)d_in[7];
    const float* W4  = (const float*)d_in[8];
    const float* b4  = (const float*)d_in[9];

    // d_out: [x_recon f32 B*IN][z f32 B*L*D][z_q f32 B*L*D][indices i32 B*L]
    float* o_xrec = (float*)d_out;
    float* o_z    = o_xrec + (size_t)BB * IN_DIM;
    float* o_zq   = o_z    + (size_t)BB * LL * DD;
    int*   o_idx  = (int*)(o_zq + (size_t)BB * LL * DD);

    // workspace carve-up (bf16 operands + intermediates), ~168 MB
    char* w = (char*)d_ws; size_t off = 0;
    u16*  xb   = (u16*)(w + off); off = alignup(off + (size_t)BB * IN_DIM * 2);
    u16*  W1t  = (u16*)(w + off); off = alignup(off + (size_t)HID * IN_DIM * 2);
    u16*  W2t  = (u16*)(w + off); off = alignup(off + (size_t)HID * HID * 2);
    u16*  W3t  = (u16*)(w + off); off = alignup(off + (size_t)HID * HID * 2);
    u16*  W4t  = (u16*)(w + off); off = alignup(off + (size_t)IN_DIM * HID * 2);
    u16*  embb = (u16*)(w + off); off = alignup(off + (size_t)KCODES * DD * 2);
    float* enr = (float*)(w + off); off = alignup(off + (size_t)KCODES * 4);
    u16*  hb   = (u16*)(w + off); off = alignup(off + (size_t)BB * HID * 2);
    u16*  zbw  = (u16*)(w + off); off = alignup(off + (size_t)BB * HID * 2);
    u16*  zqb  = (u16*)(w + off); off = alignup(off + (size_t)BB * HID * 2);
    u16*  h2b  = (u16*)(w + off); off = alignup(off + (size_t)BB * HID * 2);
    (void)ws_size; (void)in_sizes; (void)n_in; (void)out_size;

    // 1) operand preparation
    {
        int n4 = (BB * IN_DIM) / 4;
        cvt_f32_bf16<<<(n4 + 255) / 256, 256, 0, stream>>>(x, xb, n4);
        n4 = (KCODES * DD) / 4;
        cvt_f32_bf16<<<(n4 + 255) / 256, 256, 0, stream>>>(emb, embb, n4);
    }
    {
        dim3 blk(32, 8);
        cvt_transpose_bf16<<<dim3(HID / 32,  IN_DIM / 32), blk, 0, stream>>>(W1, W1t, IN_DIM, HID);
        cvt_transpose_bf16<<<dim3(HID / 32,  HID / 32),    blk, 0, stream>>>(W2, W2t, HID, HID);
        cvt_transpose_bf16<<<dim3(HID / 32,  HID / 32),    blk, 0, stream>>>(W3, W3t, HID, HID);
        cvt_transpose_bf16<<<dim3(IN_DIM/32, HID / 32),    blk, 0, stream>>>(W4, W4t, HID, IN_DIM);
    }
    emb_norm_kernel<<<2, 256, 0, stream>>>(emb, enr);

    // 2) encoder: h = relu(x@W1+b1)  [bf16 out only]
    gemm_bf16<<<dim3(HID / TN, BB / TM), 256, 0, stream>>>(
        xb, W1t, b1, nullptr, hb, BB, HID, IN_DIM, 1);

    // 3) z = h@W2+b2  -> f32 to d_out (z output) + bf16 for VQ
    gemm_bf16<<<dim3(HID / TN, BB / TM), 256, 0, stream>>>(
        hb, W2t, b2, o_z, zbw, BB, HID, HID, 0);

    // 4) vector quantization: indices + z_q (f32 out + bf16 for decoder)
    vq_kernel<<<RTOT / 128, 256, 0, stream>>>(zbw, embb, emb, enr, o_idx, o_zq, zqb);

    // 5) decoder: h2 = relu(z_q@W3+b3)  [bf16 out only]
    gemm_bf16<<<dim3(HID / TN, BB / TM), 256, 0, stream>>>(
        zqb, W3t, b3, nullptr, h2b, BB, HID, HID, 1);

    // 6) x_recon = h2@W4+b4  -> f32 to d_out
    gemm_bf16<<<dim3(IN_DIM / TN, BB / TM), 256, 0, stream>>>(
        h2b, W4t, b4, o_xrec, nullptr, BB, IN_DIM, HID, 0);
}
